// Twostage_Attention_19258633355749
// MI455X (gfx1250) — matrix operand order repaired and checked by reading, not compiled, not run on hardware
//
#include <hip/hip_runtime.h>

#define HH 768
#define BB 16
#define LL 1024
#define BL (BB * LL) // 16384

typedef __bf16 bf16;
typedef __attribute__((ext_vector_type(16))) __bf16 v16bf;
typedef __attribute__((ext_vector_type(8)))  __bf16 v8bf;
typedef __attribute__((ext_vector_type(8)))  float  v8f;
typedef int v4i_vs __attribute__((vector_size(16))); // matches builtin param: int __vector(4)

#define AS1 __attribute__((address_space(1)))
#define AS3 __attribute__((address_space(3)))

union FragU { v16bf v; v8bf h[2]; };

// ---- WMMA fragment loaders (CDNA5 16x16x32 bf16 layouts, ISA 7.12.2) ----
// A (16x32, MxK): lane l -> row (l&15); halfs 0..7 = K {kh..kh+7}, halfs 8..15 = K {16+kh..}, kh = 8*(l>>4)
__device__ __forceinline__ v16bf load_fragA(const bf16* lds, int rowBase, int lane) {
  const int r  = rowBase + (lane & 15);
  const int kh = (lane >> 4) << 3;
  FragU u;
  u.h[0] = *(const v8bf*)(lds + r * 32 + kh);
  u.h[1] = *(const v8bf*)(lds + r * 32 + 16 + kh);
  return u.v;
}
// B (32x16, KxN): lane l -> column (l&15); halfs 0..7 = K {kh..kh+7}, halfs 8..15 = K {16+kh..}, kh = 8*(l>>4)
// ldsT is the tile stored column-major: ldsT[n][k], n-stride 32
__device__ __forceinline__ v16bf load_fragB(const bf16* ldsT, int colBase, int lane) {
  const int c  = colBase + (lane & 15);
  const int k0 = (lane >> 4) << 3;
  FragU u;
  u.h[0] = *(const v8bf*)(ldsT + c * 32 + k0);
  u.h[1] = *(const v8bf*)(ldsT + c * 32 + 16 + k0);
  return u.v;
}

// ---- async global->LDS copy of one 128x32 bf16 tile (256 threads, 2x b128/thread) ----
__device__ __forceinline__ void async_b128(const bf16* g, bf16* l) {
  __builtin_amdgcn_global_load_async_to_lds_b128(
      (AS1 v4i_vs*)g, (AS3 v4i_vs*)l, 0, 0);
}
__device__ __forceinline__ void async_stage(bf16* dst /*LDS base, row stride 32*/,
                                            const bf16* src, int stride, int t) {
  const int row = t >> 1, cs = (t & 1) << 4;
  const bf16* s = src + (size_t)row * stride + cs;
  async_b128(s,     dst + row * 32 + cs);
  async_b128(s + 8, dst + row * 32 + cs + 8);
}

// ---- synchronous LDS staging helpers (256 threads stage a 128x32 tile) ----
__device__ __forceinline__ void stage_f32(bf16 (*dst)[32], const float* src, int stride, int t) {
  const int row = t >> 1, cs = (t & 1) << 4;
  const float4* s4 = (const float4*)(src + (size_t)row * stride + cs);
  float4 f0 = s4[0], f1 = s4[1], f2 = s4[2], f3 = s4[3];
  v8bf o0, o1;
  o0[0]=(bf16)f0.x; o0[1]=(bf16)f0.y; o0[2]=(bf16)f0.z; o0[3]=(bf16)f0.w;
  o0[4]=(bf16)f1.x; o0[5]=(bf16)f1.y; o0[6]=(bf16)f1.z; o0[7]=(bf16)f1.w;
  o1[0]=(bf16)f2.x; o1[1]=(bf16)f2.y; o1[2]=(bf16)f2.z; o1[3]=(bf16)f2.w;
  o1[4]=(bf16)f3.x; o1[5]=(bf16)f3.y; o1[6]=(bf16)f3.z; o1[7]=(bf16)f3.w;
  *(v8bf*)(&dst[row][cs])     = o0;
  *(v8bf*)(&dst[row][cs + 8]) = o1;
}
// transpose-stage a [32(K) x 128(N)] row-major tile into dstT[n][k]
__device__ __forceinline__ void stage_f32_T(bf16 (*dstT)[32], const float* src, int stride, int t) {
  const int kr = t & 31, ns = (t >> 5) << 4;
  const float4* s4 = (const float4*)(src + (size_t)kr * stride + ns);
  float4 f[4] = { s4[0], s4[1], s4[2], s4[3] };
  const float* fp = (const float*)f;
  #pragma unroll
  for (int i = 0; i < 16; ++i) dstT[ns + i][kr] = (bf16)fp[i];
}
__device__ __forceinline__ void stage_bf_T(bf16 (*dstT)[32], const bf16* src, int stride, int t) {
  const int kr = t & 31, ns = (t >> 5) << 4;
  const bf16* s = src + (size_t)kr * stride + ns;
  v8bf a = *(const v8bf*)(s), b = *(const v8bf*)(s + 8);
  #pragma unroll
  for (int i = 0; i < 8; ++i) { dstT[ns + i][kr] = a[i]; dstT[ns + 8 + i][kr] = b[i]; }
}

#define WMMA_BF16(A, Bx, C) \
  __builtin_amdgcn_wmma_f32_16x16x32_bf16(false, (A), false, (Bx), (short)0, (C), false, false)

// ---- K0: fold W2 = Wt @ Wo, c = bt . Wo + bo ----
__global__ void fold_kernel(const float* __restrict__ Wt, const float* __restrict__ Wo,
                            const float* __restrict__ bt, const float* __restrict__ bo,
                            float* __restrict__ W2, float* __restrict__ cb) {
  const int i = blockIdx.x * 256 + threadIdx.x;
  if (i < HH) {
    float s = 0.f;
    for (int j = 0; j < HH; ++j) s += Wt[(size_t)i * HH + j] * Wo[j];
    W2[i] = s;
  }
  if (i == 0) {
    float s = 0.f;
    for (int j = 0; j < HH; ++j) s += bt[j] * Wo[j];
    *cb = s + bo[0];
  }
}

// ---- K1: projection GEMM: out_bf16[BL,H] = X[BL,H] @ W[H,H] + bias ----
__global__ void __launch_bounds__(256) proj_kernel(const float* __restrict__ X,
    const float* __restrict__ W, const float* __restrict__ bias, bf16* __restrict__ out) {
  __shared__ bf16 As[128][32];
  __shared__ bf16 Bs[128][32];
  const int t = threadIdx.x, lane = t & 31, wid = t >> 5;
  const int wm = wid >> 1, wn = wid & 1;
  const int m0 = blockIdx.y * 128, n0 = blockIdx.x * 128;
  v8f acc[2][4] = {};
  for (int kk = 0; kk < HH; kk += 32) {
    __syncthreads();
    stage_f32(As, X + (size_t)m0 * HH + kk, HH, t);
    stage_f32_T(Bs, W + (size_t)kk * HH + n0, HH, t);
    if (kk + 32 < HH) {
      __builtin_prefetch(X + (size_t)m0 * HH + kk + 32 + (t >> 1) * HH, 0, 1);
      __builtin_prefetch(W + (size_t)(kk + 32 + (t & 31)) * HH + n0, 0, 1);
    }
    __syncthreads();
    v16bf a[2], b[4];
    a[0] = load_fragA(&As[0][0], wm * 32, lane);
    a[1] = load_fragA(&As[0][0], wm * 32 + 16, lane);
    #pragma unroll
    for (int j = 0; j < 4; ++j) b[j] = load_fragB(&Bs[0][0], wn * 64 + j * 16, lane);
    #pragma unroll
    for (int i = 0; i < 2; ++i)
      #pragma unroll
      for (int j = 0; j < 4; ++j)
        acc[i][j] = WMMA_BF16(a[i], b[j], acc[i][j]);
  }
  #pragma unroll
  for (int i = 0; i < 2; ++i)
    #pragma unroll
    for (int j = 0; j < 4; ++j) {
      const int gr = m0 + wm * 32 + i * 16 + ((lane >> 4) << 3);
      const int gc = n0 + wn * 64 + j * 16 + (lane & 15);
      const float bia = bias[gc];
      #pragma unroll
      for (int v = 0; v < 8; ++v)
        out[(size_t)(gr + v) * HH + gc] = (bf16)(acc[i][j][v] + bia);
    }
}

// ---- K2: S = q k^T * scale; P = exp(S) (bf16); Z[b,k] += column sums ----
// Double-buffered: both operands streamed with global_load_async_to_lds_b128.
__global__ void __launch_bounds__(256) scores_kernel(const bf16* __restrict__ qb,
    const bf16* __restrict__ kb, bf16* __restrict__ P, float* __restrict__ Z) {
  __shared__ bf16 As[2][128][32];
  __shared__ bf16 Bs[2][128][32];
  const int t = threadIdx.x, lane = t & 31, wid = t >> 5;
  const int wm = wid >> 1, wn = wid & 1;
  const int b = blockIdx.z;
  const int m0 = blockIdx.y * 128, n0 = blockIdx.x * 128;
  const bf16* q = qb + (size_t)b * LL * HH + (size_t)m0 * HH;
  const bf16* k = kb + (size_t)b * LL * HH + (size_t)n0 * HH; // B = k^T: row-major k IS col-major for fragB
  v8f acc[2][4] = {};
  const int NIT = HH / 32;
  async_stage(&As[0][0][0], q, HH, t);
  async_stage(&Bs[0][0][0], k, HH, t);
  for (int idx = 0; idx < NIT; ++idx) {
    const int cur = idx & 1, nxt = cur ^ 1;
    __syncthreads(); // prev iteration's readers of buffer 'nxt' are done
    if (idx + 1 < NIT) {
      async_stage(&As[nxt][0][0], q + (idx + 1) * 32, HH, t);
      async_stage(&Bs[nxt][0][0], k + (idx + 1) * 32, HH, t);
      asm volatile("s_wait_asynccnt 0x4" ::: "memory"); // cur tile's 4 copies landed (in-order)
    } else {
      asm volatile("s_wait_asynccnt 0x0" ::: "memory");
    }
    __syncthreads();
    v16bf a[2], bx[4];
    a[0] = load_fragA(&As[cur][0][0], wm * 32, lane);
    a[1] = load_fragA(&As[cur][0][0], wm * 32 + 16, lane);
    #pragma unroll
    for (int j = 0; j < 4; ++j) bx[j] = load_fragB(&Bs[cur][0][0], wn * 64 + j * 16, lane);
    #pragma unroll
    for (int i = 0; i < 2; ++i)
      #pragma unroll
      for (int j = 0; j < 4; ++j)
        acc[i][j] = WMMA_BF16(a[i], bx[j], acc[i][j]);
  }
  const float scale = 0.03608439182435161f; // 1/sqrt(768)
  bf16* Pb = P + ((size_t)b << 20);
  float* Zb = Z + (size_t)b * LL;
  #pragma unroll
  for (int i = 0; i < 2; ++i)
    #pragma unroll
    for (int j = 0; j < 4; ++j) {
      const int gr = m0 + wm * 32 + i * 16 + ((lane >> 4) << 3);
      const int gc = n0 + wn * 64 + j * 16 + (lane & 15);
      float csum = 0.f;
      #pragma unroll
      for (int v = 0; v < 8; ++v) {
        float p = __expf(acc[i][j][v] * scale);
        csum += p;
        Pb[(size_t)(gr + v) * LL + gc] = (bf16)p;
      }
      unsafeAtomicAdd(&Zb[gc], csum); // softmax over QUERY axis -> per-key column sum
    }
}

// ---- K3: vb[b,k,h] /= Z[b,k] (in place; vb is rebuilt each launch) ----
__global__ void scale_v_kernel(bf16* __restrict__ vb, const float* __restrict__ Z) {
  const size_t idx = (size_t)blockIdx.x * 256 + threadIdx.x; // < BL*HH
  const int row = (int)(idx / HH);
  vb[idx] = (bf16)((float)vb[idx] / Z[row]);
}

// ---- K4: ctx = P @ Vscaled ----
// A (P tile) async double-buffered; B (v tile) transposed synchronously in the barrier window.
__global__ void __launch_bounds__(256) ctx_kernel(const bf16* __restrict__ P,
    const bf16* __restrict__ vs, bf16* __restrict__ ctx) {
  __shared__ bf16 As[2][128][32];
  __shared__ bf16 BsT[128][32];
  const int t = threadIdx.x, lane = t & 31, wid = t >> 5;
  const int wm = wid >> 1, wn = wid & 1;
  const int b = blockIdx.z;
  const int m0 = blockIdx.y * 128, n0 = blockIdx.x * 128;
  const bf16* Pb = P + ((size_t)b << 20) + (size_t)m0 * LL;
  const bf16* vbb = vs + (size_t)b * LL * HH;
  v8f acc[2][4] = {};
  const int NIT = LL / 32;
  async_stage(&As[0][0][0], Pb, LL, t);
  for (int idx = 0; idx < NIT; ++idx) {
    const int cur = idx & 1, nxt = cur ^ 1;
    const int kk = idx * 32;
    __syncthreads(); // prev compute done: BsT and As[nxt] safe to overwrite
    stage_bf_T(BsT, vbb + (size_t)kk * HH + n0, HH, t);
    if (idx + 1 < NIT) {
      async_stage(&As[nxt][0][0], Pb + kk + 32, LL, t);
      asm volatile("s_wait_asynccnt 0x2" ::: "memory"); // cur A tile's 2 copies landed
      __builtin_prefetch(vbb + (size_t)(kk + 32 + (t & 31)) * HH + n0, 0, 1);
    } else {
      asm volatile("s_wait_asynccnt 0x0" ::: "memory");
    }
    __syncthreads();
    v16bf a[2], bx[4];
    a[0] = load_fragA(&As[cur][0][0], wm * 32, lane);
    a[1] = load_fragA(&As[cur][0][0], wm * 32 + 16, lane);
    #pragma unroll
    for (int j = 0; j < 4; ++j) bx[j] = load_fragB(&BsT[0][0], wn * 64 + j * 16, lane);
    #pragma unroll
    for (int i = 0; i < 2; ++i)
      #pragma unroll
      for (int j = 0; j < 4; ++j)
        acc[i][j] = WMMA_BF16(a[i], bx[j], acc[i][j]);
  }
  #pragma unroll
  for (int i = 0; i < 2; ++i)
    #pragma unroll
    for (int j = 0; j < 4; ++j) {
      const int gr = m0 + wm * 32 + i * 16 + ((lane >> 4) << 3);
      const int gc = n0 + wn * 64 + j * 16 + (lane & 15);
      #pragma unroll
      for (int v = 0; v < 8; ++v)
        ctx[(size_t)b * LL * HH + (size_t)(gr + v) * HH + gc] = (bf16)acc[i][j][v];
    }
}

// ---- K5: out[row] = ctx[row,:] . W2 + c ---- (one wave per row)
__global__ void __launch_bounds__(256) out_kernel(const bf16* __restrict__ ctx,
    const float* __restrict__ W2, const float* __restrict__ cb, float* __restrict__ out) {
  const int lane = threadIdx.x & 31, wid = threadIdx.x >> 5;
  const int row = blockIdx.x * 8 + wid;
  const bf16* cr = ctx + (size_t)row * HH;
  float s = 0.f;
  #pragma unroll
  for (int j = 0; j < HH / 32; ++j) {
    const int c = lane + 32 * j;
    s += (float)cr[c] * W2[c];
  }
  #pragma unroll
  for (int m = 16; m > 0; m >>= 1) s += __shfl_xor(s, m, 32);
  if (lane == 0) out[row] = s + cb[0];
}

extern "C" void kernel_launch(void* const* d_in, const int* in_sizes, int n_in,
                              void* d_out, int out_size, void* d_ws, size_t ws_size,
                              hipStream_t stream) {
  const float* query_x = (const float*)d_in[0];
  const float* x  = (const float*)d_in[1];
  const float* Wq = (const float*)d_in[2];
  const float* bq = (const float*)d_in[3];
  const float* Wk = (const float*)d_in[4];
  const float* bk = (const float*)d_in[5];
  const float* Wv = (const float*)d_in[6];
  const float* bv = (const float*)d_in[7];
  const float* Wt = (const float*)d_in[8];
  const float* bt = (const float*)d_in[9];
  const float* Wo = (const float*)d_in[10];
  const float* bo = (const float*)d_in[11];
  float* out = (float*)d_out;

  // workspace carve-out (~134 MB total)
  char* ws = (char*)d_ws;
  size_t off = 0;
  auto carve = [&](size_t bytes) { char* p = ws + off; off += (bytes + 255) & ~(size_t)255; return p; };
  bf16*  qb  = (bf16*)carve((size_t)BL * HH * 2);
  bf16*  kb  = (bf16*)carve((size_t)BL * HH * 2);
  bf16*  vb  = (bf16*)carve((size_t)BL * HH * 2);
  bf16*  Pm  = (bf16*)carve((size_t)BB * LL * LL * 2);
  float* Z   = (float*)carve((size_t)BB * LL * 4);
  bf16*  ctx = (bf16*)carve((size_t)BL * HH * 2);
  float* W2  = (float*)carve(HH * 4);
  float* cb  = (float*)carve(256);

  (void)hipMemsetAsync(Z, 0, (size_t)BB * LL * 4, stream);
  fold_kernel<<<3, 256, 0, stream>>>(Wt, Wo, bt, bo, W2, cb);

  dim3 gProj(HH / 128, BL / 128); // (6, 128)
  proj_kernel<<<gProj, 256, 0, stream>>>(query_x, Wq, bq, qb);
  proj_kernel<<<gProj, 256, 0, stream>>>(x,       Wk, bk, kb);
  proj_kernel<<<gProj, 256, 0, stream>>>(x,       Wv, bv, vb);

  scores_kernel<<<dim3(LL / 128, LL / 128, BB), 256, 0, stream>>>(qb, kb, Pm, Z);
  scale_v_kernel<<<(BL * HH) / 256, 256, 0, stream>>>(vb, Z);
  ctx_kernel<<<dim3(HH / 128, LL / 128, BB), 256, 0, stream>>>(Pm, vb, ctx);
  out_kernel<<<BL / 8, 256, 0, stream>>>(ctx, W2, cb, out);
}
